// MambaForecastModel_9122510536852
// MI455X (gfx1250) — matrix operand order repaired and checked by reading, not compiled, not run on hardware
//
#include <hip/hip_runtime.h>
#include <hip/hip_bf16.h>

// ---------------- model dims ----------------
#define BATCH   8
#define SEQ     512
#define NROWS   (BATCH * SEQ)      // 4096
#define IN_DIM  32
#define DMODEL  256
#define DINNER  512
#define DSTATE  16
#define DTRANK  16
#define DCONV   4
#define NLAYERS 4
#define DBL_W   (DTRANK + 2 * DSTATE)   // 48
#define DTPAD   32                      // dt_proj K padded 16 -> 32
#define LN_EPS  1e-5f

typedef __attribute__((ext_vector_type(16))) _Float16 v16h;
typedef __attribute__((ext_vector_type(8)))  _Float16 v8h;
typedef __attribute__((ext_vector_type(8)))  float    v8f;

enum { ACT_NONE = 0, ACT_RELU = 1, ACT_SILU = 2, ACT_SOFTPLUS = 3 };

// ---------------- f32 -> f16 convert ----------------
__global__ void cvt_f16_kernel(const float* __restrict__ s, _Float16* __restrict__ d, int n) {
    int i = blockIdx.x * blockDim.x + threadIdx.x;
    if (i < n) d[i] = (_Float16)s[i];
}

// pack dt_proj_w [L][512][16] -> [L][512][32] f16 with zero pad
__global__ void pack_dtw_kernel(const float* __restrict__ w, _Float16* __restrict__ o, int total) {
    int i = blockIdx.x * blockDim.x + threadIdx.x;
    if (i >= total) return;
    int c = i & (DTPAD - 1);
    int row = i >> 5;
    o[i] = (c < DTRANK) ? (_Float16)w[row * DTRANK + c] : (_Float16)0.0f;
}

// pack dt columns of dbl [4096][48] -> [4096][32] f16 with zero pad
__global__ void pack_dt_kernel(const float* __restrict__ dbl, _Float16* __restrict__ o, int total) {
    int i = blockIdx.x * blockDim.x + threadIdx.x;
    if (i >= total) return;
    int c = i & (DTPAD - 1);
    int row = i >> 5;
    o[i] = (c < DTRANK) ? (_Float16)dbl[row * DBL_W + c] : (_Float16)0.0f;
}

// ---------------- fragment helpers ----------------
// A frag (16x32): lane holds row (lane&15), K = {half*8 + 0..7} and {16 + half*8 + 0..7}
__device__ __forceinline__ v16h load_a_frag(const _Float16* base, int k) {
    v8h lo = *(const v8h*)(base + k);
    v8h hi = *(const v8h*)(base + k + 16);
    v16h f;
#pragma unroll
    for (int i = 0; i < 8; ++i) { f[i] = lo[i]; f[i + 8] = hi[i]; }
    return f;
}
// B frag (32x16): lane holds col (lane&15), K = {half*8 + 0..7} and {16 + half*8 + 0..7}
__device__ __forceinline__ v16h load_b_frag(const _Float16* base, int k) {
    v8h lo = *(const v8h*)(base + k);
    v8h hi = *(const v8h*)(base + k + 16);
    v16h f;
#pragma unroll
    for (int i = 0; i < 8; ++i) { f[i] = lo[i]; f[i + 8] = hi[i]; }
    return f;
}

// ---------------- WMMA GEMM: C[M,N] = A[M,K] @ W[N,K]^T (+bias)(+act)(+addsrc) ---------------
// A, W row-major f16; K % 32 == 0; M % 32 == 0. Each wave computes a 32x32 output
// macro-tile (2x2 of 16x16 WMMA tiles) so each fragment feeds two WMMAs.
// Block = 4 waves spanning 128 columns of N; grid = (ceil(N/128), M/32).
__global__ __launch_bounds__(128)
void gemm_f16_wmma(const _Float16* __restrict__ A, const _Float16* __restrict__ W,
                   const float* bias, const float* addsrc, float* Cout,
                   int M, int N, int K, int act) {
    const int lane = threadIdx.x & 31;
    const int wave = threadIdx.x >> 5;
    const int n0 = (blockIdx.x * 4 + wave) * 32;   // wave's 32-wide N slab
    const int m0 = blockIdx.y * 32;                // wave's 32-tall M slab
    if (n0 >= N) return;

    const int r    = lane & 15;
    const int half = lane >> 4;

    // A rows m0+r and m0+16+r (M % 32 == 0, always in range)
    const _Float16* Ab0 = A + (size_t)(m0 + r) * K + (half << 3);
    const _Float16* Ab1 = A + (size_t)(m0 + 16 + r) * K + (half << 3);
    // B cols n0+r and n0+16+r; clamp row index for ragged N (stores are guarded)
    int c0 = n0 + r;       if (c0 > N - 1) c0 = N - 1;
    int c1 = n0 + 16 + r;  if (c1 > N - 1) c1 = N - 1;
    const _Float16* Wb0 = W + (size_t)c0 * K + (half << 3);
    const _Float16* Wb1 = W + (size_t)c1 * K + (half << 3);

    v8f acc00 = {}, acc01 = {}, acc10 = {}, acc11 = {};
    for (int k = 0; k < K; k += 32) {
        if (k + 32 < K) {
            __builtin_prefetch(Ab0 + k + 32, 0, 3);
            __builtin_prefetch(Ab1 + k + 32, 0, 3);
            __builtin_prefetch(Wb0 + k + 32, 0, 3);
            __builtin_prefetch(Wb1 + k + 32, 0, 3);
        }
        v16h a0 = load_a_frag(Ab0, k);
        v16h a1 = load_a_frag(Ab1, k);
        v16h b0 = load_b_frag(Wb0, k);
        v16h b1 = load_b_frag(Wb1, k);
        acc00 = __builtin_amdgcn_wmma_f32_16x16x32_f16(false, a0, false, b0, (short)0, acc00, false, false);
        acc01 = __builtin_amdgcn_wmma_f32_16x16x32_f16(false, a0, false, b1, (short)0, acc01, false, false);
        acc10 = __builtin_amdgcn_wmma_f32_16x16x32_f16(false, a1, false, b0, (short)0, acc10, false, false);
        acc11 = __builtin_amdgcn_wmma_f32_16x16x32_f16(false, a1, false, b1, (short)0, acc11, false, false);
    }

    // epilogue: C/D layout — col = tile_n + (lane&15), row = tile_m + v + 8*half
    const int mbase = m0 + (half << 3);
#pragma unroll
    for (int ni = 0; ni < 2; ++ni) {
        const int ncol = n0 + ni * 16 + r;
        if (ncol >= N) continue;
        const float bv = bias ? bias[ncol] : 0.0f;
#pragma unroll
        for (int mi = 0; mi < 2; ++mi) {
            const v8f& acc = (mi == 0) ? (ni == 0 ? acc00 : acc01)
                                       : (ni == 0 ? acc10 : acc11);
#pragma unroll
            for (int v = 0; v < 8; ++v) {
                float val = acc[v] + bv;
                if (act == ACT_RELU) {
                    val = fmaxf(val, 0.0f);
                } else if (act == ACT_SILU) {
                    val = val / (1.0f + __expf(-val));
                } else if (act == ACT_SOFTPLUS) {
                    val = (val > 20.0f) ? val : log1pf(__expf(val));
                }
                size_t idx = (size_t)(mbase + mi * 16 + v) * N + ncol;
                if (addsrc) val += addsrc[idx];
                Cout[idx] = val;
            }
        }
    }
}

// ---------------- LayerNorm (D=256) -> f16 ----------------
__global__ __launch_bounds__(DMODEL)
void layernorm_f16_kernel(const float* __restrict__ x, const float* __restrict__ w,
                          const float* __restrict__ b, _Float16* __restrict__ o) {
    const int row = blockIdx.x;
    const int tid = threadIdx.x;
    __shared__ float s1[DMODEL];
    __shared__ float s2[DMODEL];
    float v = x[(size_t)row * DMODEL + tid];
    s1[tid] = v;
    s2[tid] = v * v;
    __syncthreads();
    for (int off = DMODEL / 2; off > 0; off >>= 1) {
        if (tid < off) { s1[tid] += s1[tid + off]; s2[tid] += s2[tid + off]; }
        __syncthreads();
    }
    float mu  = s1[0] * (1.0f / DMODEL);
    float var = s2[0] * (1.0f / DMODEL) - mu * mu;
    float rs  = rsqrtf(var + LN_EPS);
    o[(size_t)row * DMODEL + tid] = (_Float16)((v - mu) * rs * w[tid] + b[tid]);
}

// ---------------- causal depthwise conv (width 4) + SiLU on u = xz[:, :512] ----------------
__global__ void conv_silu_kernel(const float* __restrict__ xz, const float* __restrict__ cw,
                                 const float* __restrict__ cb, float* __restrict__ u32,
                                 _Float16* __restrict__ u16) {
    int idx = blockIdx.x * blockDim.x + threadIdx.x;   // over 4096*512
    if (idx >= NROWS * DINNER) return;
    int d = idx & (DINNER - 1);
    int row = idx >> 9;            // b*SEQ + t
    int t = row & (SEQ - 1);
    float acc = cb[d];
#pragma unroll
    for (int i = 0; i < DCONV; ++i) {
        int back = (DCONV - 1) - i;           // t - back
        if (t >= back)
            acc += cw[d * DCONV + i] * xz[(size_t)(row - back) * (2 * DINNER) + d];
    }
    acc = acc / (1.0f + __expf(-acc));        // SiLU
    u32[idx] = acc;
    u16[idx] = (_Float16)acc;
}

// ---------------- selective scan fused with D-skip and SiLU(z) gate; writes f16 -----------
__global__ __launch_bounds__(DINNER)
void scan_kernel(const float* __restrict__ u, const float* __restrict__ dtf,
                 const float* __restrict__ dbl, const float* __restrict__ xz,
                 const float* __restrict__ A_log, const float* __restrict__ Dp,
                 _Float16* __restrict__ yg) {
    const int b = blockIdx.x;
    const int d = threadIdx.x;
    float Av[DSTATE];
#pragma unroll
    for (int s = 0; s < DSTATE; ++s) Av[s] = -__expf(A_log[d * DSTATE + s]);
    float hs[DSTATE];
#pragma unroll
    for (int s = 0; s < DSTATE; ++s) hs[s] = 0.0f;
    const float Dd = Dp[d];
    __shared__ float sBC[2 * DSTATE];
    for (int t = 0; t < SEQ; ++t) {
        const int row = b * SEQ + t;
        __syncthreads();
        if (d < 2 * DSTATE) sBC[d] = dbl[(size_t)row * DBL_W + DTRANK + d];
        __syncthreads();
        float dt = dtf[(size_t)row * DINNER + d];
        float uu = u[(size_t)row * DINNER + d];
        float du = dt * uu;
        float y = 0.0f;
#pragma unroll
        for (int s = 0; s < DSTATE; ++s) {
            float dA = __expf(dt * Av[s]);
            hs[s] = dA * hs[s] + du * sBC[s];
            y += hs[s] * sBC[DSTATE + s];
        }
        float zv = xz[(size_t)row * (2 * DINNER) + DINNER + d];
        y = (y + uu * Dd) * (zv / (1.0f + __expf(-zv)));
        yg[(size_t)row * DINNER + d] = (_Float16)y;
    }
}

// ---------------- head (M = 8, scalar is fine) ----------------
__global__ __launch_bounds__(DMODEL)
void head1_kernel(const float* __restrict__ h, const float* __restrict__ w1,
                  const float* __restrict__ b1, float* __restrict__ o) {
    const int b = blockIdx.x;
    const int n = threadIdx.x;
    __shared__ float rowv[DMODEL];
    rowv[n] = h[(size_t)(b * SEQ + (SEQ - 1)) * DMODEL + n];
    __syncthreads();
    float acc = b1[n];
    for (int j = 0; j < DMODEL; ++j) acc += rowv[j] * w1[n * DMODEL + j];
    o[b * DMODEL + n] = fmaxf(acc, 0.0f);
}

__global__ __launch_bounds__(DMODEL)
void head2_kernel(const float* __restrict__ o1, const float* __restrict__ w2,
                  const float* __restrict__ b2, float* __restrict__ out) {
    const int b = blockIdx.x;
    const int j = threadIdx.x;
    __shared__ float red[DMODEL];
    red[j] = o1[b * DMODEL + j] * w2[j];
    __syncthreads();
    for (int off = DMODEL / 2; off > 0; off >>= 1) {
        if (j < off) red[j] += red[j + off];
        __syncthreads();
    }
    if (j == 0) out[b] = red[0] + b2[0];
}

// ---------------- host side ----------------
static inline size_t align256(size_t x) { return (x + 255) & ~(size_t)255; }
static inline int cdiv(int a, int b) { return (a + b - 1) / b; }

extern "C" void kernel_launch(void* const* d_in, const int* in_sizes, int n_in,
                              void* d_out, int out_size, void* d_ws, size_t ws_size,
                              hipStream_t stream) {
    (void)in_sizes; (void)n_in; (void)out_size; (void)ws_size;

    const float* x_in   = (const float*)d_in[0];   // (8,512,32)
    const float* ipw    = (const float*)d_in[1];   // (256,32)
    const float* ipb    = (const float*)d_in[2];   // (256)
    const float* ln_w   = (const float*)d_in[3];   // (4,256)
    const float* ln_b   = (const float*)d_in[4];   // (4,256)
    const float* inpw   = (const float*)d_in[5];   // (4,1024,256)
    const float* convw  = (const float*)d_in[6];   // (4,512,4)
    const float* convb  = (const float*)d_in[7];   // (4,512)
    const float* xpw    = (const float*)d_in[8];   // (4,48,512)
    const float* dtpw   = (const float*)d_in[9];   // (4,512,16)
    const float* dtpb   = (const float*)d_in[10];  // (4,512)
    const float* A_log  = (const float*)d_in[11];  // (4,512,16)
    const float* Dparam = (const float*)d_in[12];  // (4,512)
    const float* outw   = (const float*)d_in[13];  // (4,256,512)
    const float* hw1    = (const float*)d_in[14];  // (256,256)
    const float* hb1    = (const float*)d_in[15];  // (256)
    const float* hw2    = (const float*)d_in[16];  // (1,256)
    const float* hb2    = (const float*)d_in[17];  // (1)
    float* out = (float*)d_out;

    // workspace carve-up
    char* p = (char*)d_ws;
    auto takeH = [&](size_t n) { _Float16* q = (_Float16*)p; p += align256(n * sizeof(_Float16)); return q; };
    auto takeF = [&](size_t n) { float* q = (float*)p; p += align256(n * sizeof(float)); return q; };

    _Float16* x16    = takeH((size_t)NROWS * IN_DIM);
    _Float16* ipw16  = takeH((size_t)DMODEL * IN_DIM);
    _Float16* inw16  = takeH((size_t)NLAYERS * 2 * DINNER * DMODEL);
    _Float16* xpw16  = takeH((size_t)NLAYERS * DBL_W * DINNER);
    _Float16* dtw16  = takeH((size_t)NLAYERS * DINNER * DTPAD);
    _Float16* outw16 = takeH((size_t)NLAYERS * DMODEL * DINNER);
    float*    hbuf   = takeF((size_t)NROWS * DMODEL);
    _Float16* ln16   = takeH((size_t)NROWS * DMODEL);
    float*    xz     = takeF((size_t)NROWS * 2 * DINNER);
    float*    u32    = takeF((size_t)NROWS * DINNER);
    _Float16* u16    = takeH((size_t)NROWS * DINNER);
    float*    dbl    = takeF((size_t)NROWS * DBL_W);
    _Float16* dt16   = takeH((size_t)NROWS * DTPAD);
    float*    dtf    = takeF((size_t)NROWS * DINNER);
    _Float16* yg16   = takeH((size_t)NROWS * DINNER);
    float*    hl1    = takeF((size_t)BATCH * DMODEL);

    const int TB = 256;
    // per-call f16 conversions (deterministic, no caching)
    {
        int n;
        n = NROWS * IN_DIM;
        cvt_f16_kernel<<<cdiv(n, TB), TB, 0, stream>>>(x_in, x16, n);
        n = DMODEL * IN_DIM;
        cvt_f16_kernel<<<cdiv(n, TB), TB, 0, stream>>>(ipw, ipw16, n);
        n = NLAYERS * 2 * DINNER * DMODEL;
        cvt_f16_kernel<<<cdiv(n, TB), TB, 0, stream>>>(inpw, inw16, n);
        n = NLAYERS * DBL_W * DINNER;
        cvt_f16_kernel<<<cdiv(n, TB), TB, 0, stream>>>(xpw, xpw16, n);
        n = NLAYERS * DMODEL * DINNER;
        cvt_f16_kernel<<<cdiv(n, TB), TB, 0, stream>>>(outw, outw16, n);
        n = NLAYERS * DINNER * DTPAD;
        pack_dtw_kernel<<<cdiv(n, TB), TB, 0, stream>>>(dtpw, dtw16, n);
    }

    // input projection: h = x @ ipw^T + ipb   (M=4096, N=256, K=32)
    {
        dim3 g(cdiv(DMODEL, 128), NROWS / 32);
        gemm_f16_wmma<<<g, 128, 0, stream>>>(x16, ipw16, ipb, nullptr, hbuf,
                                             NROWS, DMODEL, IN_DIM, ACT_NONE);
    }

    for (int l = 0; l < NLAYERS; ++l) {
        const float* lnw_l  = ln_w  + l * DMODEL;
        const float* lnb_l  = ln_b  + l * DMODEL;
        const _Float16* inw_l  = inw16  + (size_t)l * 2 * DINNER * DMODEL;
        const float* cw_l   = convw + (size_t)l * DINNER * DCONV;
        const float* cb_l   = convb + (size_t)l * DINNER;
        const _Float16* xpw_l  = xpw16  + (size_t)l * DBL_W * DINNER;
        const _Float16* dtw_l  = dtw16  + (size_t)l * DINNER * DTPAD;
        const float* dtpb_l = dtpb  + (size_t)l * DINNER;
        const float* Alog_l = A_log + (size_t)l * DINNER * DSTATE;
        const float* D_l    = Dparam + (size_t)l * DINNER;
        const _Float16* outw_l = outw16 + (size_t)l * DMODEL * DINNER;

        // 1) LayerNorm -> f16
        layernorm_f16_kernel<<<NROWS, DMODEL, 0, stream>>>(hbuf, lnw_l, lnb_l, ln16);

        // 2) xz = ln @ in_proj^T   (M=4096, N=1024, K=256)
        {
            dim3 g(cdiv(2 * DINNER, 128), NROWS / 32);
            gemm_f16_wmma<<<g, 128, 0, stream>>>(ln16, inw_l, nullptr, nullptr, xz,
                                                 NROWS, 2 * DINNER, DMODEL, ACT_NONE);
        }

        // 3) u = silu(causal_dw_conv(xz[:, :512]))
        conv_silu_kernel<<<cdiv(NROWS * DINNER, TB), TB, 0, stream>>>(xz, cw_l, cb_l, u32, u16);

        // 4) dbl = u @ x_proj^T    (M=4096, N=48, K=512)
        {
            dim3 g(cdiv(DBL_W, 128), NROWS / 32);
            gemm_f16_wmma<<<g, 128, 0, stream>>>(u16, xpw_l, nullptr, nullptr, dbl,
                                                 NROWS, DBL_W, DINNER, ACT_NONE);
        }

        // 5) pack dt (pad K 16->32)
        pack_dt_kernel<<<cdiv(NROWS * DTPAD, TB), TB, 0, stream>>>(dbl, dt16, NROWS * DTPAD);

        // 6) dtf = softplus(dt @ dt_proj^T + dtpb)   (M=4096, N=512, K=32)
        {
            dim3 g(cdiv(DINNER, 128), NROWS / 32);
            gemm_f16_wmma<<<g, 128, 0, stream>>>(dt16, dtw_l, dtpb_l, nullptr, dtf,
                                                 NROWS, DINNER, DTPAD, ACT_SOFTPLUS);
        }

        // 7) selective scan fused with D-skip + SiLU(z) gate -> yg16
        scan_kernel<<<BATCH, DINNER, 0, stream>>>(u32, dtf, dbl, xz, Alog_l, D_l, yg16);

        // 8) h += yg @ out_proj^T  (M=4096, N=256, K=512), fused residual add
        {
            dim3 g(cdiv(DMODEL, 128), NROWS / 32);
            gemm_f16_wmma<<<g, 128, 0, stream>>>(yg16, outw_l, nullptr, hbuf, hbuf,
                                                 NROWS, DMODEL, DINNER, ACT_NONE);
        }
    }

    // head
    head1_kernel<<<BATCH, DMODEL, 0, stream>>>(hbuf, hw1, hb1, hl1);
    head2_kernel<<<BATCH, DMODEL, 0, stream>>>(hl1, hw2, hb2, out);
}
